// MHRPRChunkAttentionLayer_23596550324713
// MI455X (gfx1250) — matrix operand order repaired and checked by reading, not compiled, not run on hardware
//
#include <hip/hip_runtime.h>
#include <math.h>

#define BB   4
#define DD   512
#define TT   8192
#define HH   8
#define CHK  64
#define NTOK (BB*TT)     /* 32768 tokens */
#define HD   64

typedef __bf16 bf16;
typedef __bf16 v16bf  __attribute__((ext_vector_type(16)));
typedef __bf16 bf16x4 __attribute__((ext_vector_type(4)));
typedef float  v8f    __attribute__((ext_vector_type(8)));
typedef unsigned int u32x4 __attribute__((ext_vector_type(4)));

struct U32x8 { u32x4 a, b; };

// ---- bf16 <-> f32 via explicit RNE bit math ----
static __device__ __forceinline__ bf16 f2bf(float f) {
    unsigned int u = __builtin_bit_cast(unsigned int, f);
    u = (u + 0x7FFFu + ((u >> 16) & 1u)) >> 16;
    unsigned short s = (unsigned short)u;
    return __builtin_bit_cast(bf16, s);
}
static __device__ __forceinline__ float bf2f(bf16 h) {
    unsigned short s = __builtin_bit_cast(unsigned short, h);
    unsigned int u = ((unsigned int)s) << 16;
    return __builtin_bit_cast(float, u);
}

// ---- WMMA fragment loaders (layouts per CDNA5 ISA 7.12.2) ----
// A (16x32 bf16): lane L holds row M = L&15; K window base (L>>4)*8, elems [0..7]=K..K+7, [8..15]=K+16..K+23
static __device__ __forceinline__ v16bf frag_a_bf16(const bf16* base, int ld) {
    int lane = threadIdx.x & 31;
    const bf16* p = base + (size_t)(lane & 15) * ld + (lane >> 4) * 8;
    U32x8 u;
    u.a = *(const u32x4*)(p);
    u.b = *(const u32x4*)(p + 16);
    return __builtin_bit_cast(v16bf, u);
}
// B (32x16 bf16): lane L holds column N = L&15; K window base (L>>4)*8, elems [0..7]=K..K+7, [8..15]=K+16..K+23
// Source matrix is "N-major" (row n contiguous in K): a column of B is a row of the source.
static __device__ __forceinline__ v16bf frag_b_bf16(const bf16* base, int ld) {
    int lane = threadIdx.x & 31;
    const bf16* p = base + (size_t)(lane & 15) * ld + (lane >> 4) * 8;
    U32x8 u;
    u.a = *(const u32x4*)(p);
    u.b = *(const u32x4*)(p + 16);
    return __builtin_bit_cast(v16bf, u);
}
static __device__ __forceinline__ v8f wmma_bf16(v16bf a, v16bf b, v8f c) {
    return __builtin_amdgcn_wmma_f32_16x16x32_bf16(false, a, false, b, (short)0, c, false, false);
}

// ---- Kernel W: fp32 weight matrix (DD*DD) -> bf16, one-time ----
__global__ __launch_bounds__(256) void k_cvt_w(const float* __restrict__ src,
                                               bf16* __restrict__ dst) {
    size_t i4 = ((size_t)blockIdx.x * 256 + threadIdx.x) * 4;
    float4 v = *(const float4*)(src + i4);
    bf16x4 o;
    o[0] = f2bf(v.x); o[1] = f2bf(v.y); o[2] = f2bf(v.z); o[3] = f2bf(v.w);
    *(bf16x4*)(dst + i4) = o;
}

// ---- Kernel 0: (B, D, T) fp32 -> token-major bf16 (NTOK, D) through a 32x32 LDS tile ----
__global__ __launch_bounds__(256) void k_transpose_cvt(const float* __restrict__ src,
                                                       bf16* __restrict__ dst) {
    __shared__ float tile[32][33];
    int b  = blockIdx.z;
    int d0 = blockIdx.y * 32;
    int t0 = blockIdx.x * 32;
    int tid = threadIdx.x;
    int row = tid >> 3;        // 0..31
    int c4  = (tid & 7) * 4;   // 0,4,...,28
    float4 v = *(const float4*)(src + ((size_t)(b * DD + d0 + row) * TT + t0 + c4));
    tile[row][c4 + 0] = v.x; tile[row][c4 + 1] = v.y;
    tile[row][c4 + 2] = v.z; tile[row][c4 + 3] = v.w;
    __syncthreads();
    bf16x4 o;
    o[0] = f2bf(tile[c4 + 0][row]);
    o[1] = f2bf(tile[c4 + 1][row]);
    o[2] = f2bf(tile[c4 + 2][row]);
    o[3] = f2bf(tile[c4 + 3][row]);
    *(bf16x4*)(dst + ((size_t)(b * TT + t0 + row) * DD + d0 + c4)) = o;
}

// ---- Kernel 1: token-major projection  out[t,j] = sum_d X[t,d]*Wb[j,d] + bias[j]  (bf16 out) ----
// Block 256 thr = 8 waves; block tile 128(M=token) x 128(N=feat); wave tile 32x64 = 2x4 WMMA tiles.
__global__ __launch_bounds__(256) void k_gemm_tok(const bf16* __restrict__ X,
                                                  const bf16* __restrict__ Wb,
                                                  const float* __restrict__ bias,
                                                  bf16* __restrict__ out) {
    int wave = threadIdx.x >> 5;
    int wm = wave >> 1, wn = wave & 1;
    int gm0 = blockIdx.x * 128 + wm * 32;
    int gn0 = blockIdx.y * 128 + wn * 64;
    v8f acc[2][4] = {};
    for (int kk = 0; kk < DD; kk += 32) {
        v16bf a0 = frag_a_bf16(X + (size_t)gm0 * DD + kk, DD);
        v16bf a1 = frag_a_bf16(X + (size_t)(gm0 + 16) * DD + kk, DD);
        if (kk + 32 < DD)
            __builtin_prefetch(X + (size_t)(gm0 + (threadIdx.x & 31)) * DD + kk + 32, 0, 0);
#pragma unroll
        for (int j = 0; j < 4; ++j) {
            v16bf b = frag_b_bf16(Wb + (size_t)(gn0 + j * 16) * DD + kk, DD);
            acc[0][j] = wmma_bf16(a0, b, acc[0][j]);
            acc[1][j] = wmma_bf16(a1, b, acc[1][j]);
        }
    }
    int lane = threadIdx.x & 31, n15 = lane & 15, hi = lane >> 4;
#pragma unroll
    for (int i = 0; i < 2; ++i)
#pragma unroll
        for (int j = 0; j < 4; ++j) {
            int gn = gn0 + j * 16 + n15;
            float bj = bias[gn];
#pragma unroll
            for (int r = 0; r < 8; ++r) {
                int gm = gm0 + i * 16 + r + hi * 8;
                out[(size_t)gm * DD + gn] = f2bf(acc[i][j][r] + bj);
            }
        }
}

// ---- Kernel 2: feature-major projection  out[j,t] = sum_d Wb[j,d]*X[t,d] + bias[j] ----
// FINAL=false: bf16 (D, NTOK).  FINAL=true: fp32 (B, D, T) with mask multiply.
template <bool FINAL>
__global__ __launch_bounds__(256) void k_gemm_feat(const bf16* __restrict__ Wb,
                                                   const bf16* __restrict__ X,
                                                   const float* __restrict__ bias,
                                                   const float* __restrict__ masks,
                                                   bf16* __restrict__ outb,
                                                   float* __restrict__ outf) {
    int wave = threadIdx.x >> 5;
    int wm = wave >> 1, wn = wave & 1;
    int gm0 = blockIdx.x * 128 + wm * 32;   // feature
    int gn0 = blockIdx.y * 128 + wn * 64;   // token
    v8f acc[2][4] = {};
    for (int kk = 0; kk < DD; kk += 32) {
        v16bf a0 = frag_a_bf16(Wb + (size_t)gm0 * DD + kk, DD);
        v16bf a1 = frag_a_bf16(Wb + (size_t)(gm0 + 16) * DD + kk, DD);
        if (kk + 32 < DD)
            __builtin_prefetch(X + (size_t)(gn0 + (threadIdx.x & 31)) * DD + kk + 32, 0, 0);
#pragma unroll
        for (int j = 0; j < 4; ++j) {
            v16bf b = frag_b_bf16(X + (size_t)(gn0 + j * 16) * DD + kk, DD);
            acc[0][j] = wmma_bf16(a0, b, acc[0][j]);
            acc[1][j] = wmma_bf16(a1, b, acc[1][j]);
        }
    }
    int lane = threadIdx.x & 31, n15 = lane & 15, hi = lane >> 4;
#pragma unroll
    for (int i = 0; i < 2; ++i)
#pragma unroll
        for (int j = 0; j < 4; ++j) {
            int gn = gn0 + j * 16 + n15;  // token
#pragma unroll
            for (int r = 0; r < 8; ++r) {
                int gm = gm0 + i * 16 + r + hi * 8;  // feature
                float v = acc[i][j][r] + bias[gm];
                if (FINAL) {
                    int b = gn >> 13;        // / TT
                    int t = gn & (TT - 1);
                    outf[((size_t)b * DD + gm) * TT + t] = v * masks[gn];
                } else {
                    outb[(size_t)gm * NTOK + gn] = f2bf(v);
                }
            }
        }
}

// ---- Kernel 3: fused chunk attention.  One block (4 waves) per (chunk, head). ----
__global__ __launch_bounds__(128) void k_attn(const bf16* __restrict__ qp,   // (NTOK, D) token-major
                                              const bf16* __restrict__ kp,   // (NTOK, D) token-major
                                              const bf16* __restrict__ vpT,  // (D, NTOK) feature-major
                                              const float* __restrict__ masks,
                                              const float* __restrict__ relk,  // (5, 64)
                                              const float* __restrict__ relv,  // (5, 64)
                                              bf16* __restrict__ xo) {       // (NTOK, D) token-major
    __shared__ float rkS[5 * 64];
    __shared__ float rvS[5 * 64];
    __shared__ float rQ[64 * 5];   // rQ[i][j] = dot(q_i, rel_k[j])
    __shared__ float sJ[64 * 5];   // per-row probability mass in each rel bucket
    __shared__ float mS[64];
    __shared__ float msum;
    __shared__ bf16 P[64 * 72];    // probabilities, pitch 72 keeps 16B alignment

    int tid = threadIdx.x;
    int cg = blockIdx.x;   // chunk 0..511
    int h  = blockIdx.y;   // head 0..7
    int tbase = cg * CHK;

    for (int idx = tid; idx < 320; idx += 128) { rkS[idx] = relk[idx]; rvS[idx] = relv[idx]; }
    if (tid < 64) mS[tid] = masks[tbase + tid];
    __syncthreads();
    if (tid == 0) { float s = 0.f; for (int kx = 0; kx < 64; ++kx) s += mS[kx]; msum = s; }
    if (tid < 64) {
        int i = tid;
        const bf16* qrow = qp + (size_t)(tbase + i) * DD + h * HD;
        float a0 = 0, a1 = 0, a2 = 0, a3 = 0, a4 = 0;
        for (int e = 0; e < HD; ++e) {
            float qv = bf2f(qrow[e]);
            a0 += qv * rkS[e];        a1 += qv * rkS[64 + e];
            a2 += qv * rkS[128 + e];  a3 += qv * rkS[192 + e];
            a4 += qv * rkS[256 + e];
        }
        rQ[i * 5 + 0] = a0; rQ[i * 5 + 1] = a1; rQ[i * 5 + 2] = a2;
        rQ[i * 5 + 3] = a3; rQ[i * 5 + 4] = a4;
    }
    __syncthreads();
    bool anymask = (msum != 0.0f);  // all-zero chunks are unmasked per reference

    int wave = tid >> 5, lane = tid & 31, n15 = lane & 15, hi = lane >> 4;
    int i0 = wave * 16;

    // S = Q Kh^T : wave owns 16(i) x 64(k)
    v8f c[4] = {};
    for (int e0 = 0; e0 < HD; e0 += 32) {
        v16bf a = frag_a_bf16(qp + (size_t)(tbase + i0) * DD + h * HD + e0, DD);
#pragma unroll
        for (int ct = 0; ct < 4; ++ct) {
            v16bf b = frag_b_bf16(kp + (size_t)(tbase + ct * 16) * DD + h * HD + e0, DD);
            c[ct] = wmma_bf16(a, b, c[ct]);
        }
    }
    // rel-K term, scale, mask, row softmax (row spread over 16 lanes of a half + 4 tiles)
    float sv[4][8];
#pragma unroll
    for (int ct = 0; ct < 4; ++ct)
#pragma unroll
        for (int r = 0; r < 8; ++r) {
            int i = i0 + r + hi * 8;
            int kx = ct * 16 + n15;
            int dk = kx - i; dk = dk < -2 ? -2 : (dk > 2 ? 2 : dk);
            float val = (c[ct][r] + rQ[i * 5 + dk + 2]) * 0.125f;  // / sqrt(64)
            if (anymask && mS[kx] == 0.0f) val = -1e30f;
            sv[ct][r] = val;
        }
#pragma unroll
    for (int r = 0; r < 8; ++r) {
        float mx = fmaxf(fmaxf(sv[0][r], sv[1][r]), fmaxf(sv[2][r], sv[3][r]));
        for (int m = 1; m < 16; m <<= 1) mx = fmaxf(mx, __shfl_xor(mx, m, 32));
        float sum = 0.f;
#pragma unroll
        for (int ct = 0; ct < 4; ++ct) { float e = __expf(sv[ct][r] - mx); sv[ct][r] = e; sum += e; }
        for (int m = 1; m < 16; m <<= 1) sum += __shfl_xor(sum, m, 32);
        float inv = 1.0f / sum;
        int i = i0 + r + hi * 8;
#pragma unroll
        for (int ct = 0; ct < 4; ++ct) P[i * 72 + ct * 16 + n15] = f2bf(sv[ct][r] * inv);
    }
    __syncthreads();
    // bucket sums for rel-V:  j=0: k<=i-2, j=1: k=i-1, j=2: k=i, j=3: k=i+1, j=4: k>=i+2
    if (tid < 64) {
        int i = tid;
        const bf16* prow = P + i * 72;
        float s0 = 0.f, s4 = 0.f;
        for (int kx = 0; kx <= i - 2; ++kx) s0 += bf2f(prow[kx]);
        for (int kx = i + 2; kx < 64; ++kx) s4 += bf2f(prow[kx]);
        sJ[i * 5 + 0] = s0;
        sJ[i * 5 + 1] = (i >= 1) ? bf2f(prow[i - 1]) : 0.f;
        sJ[i * 5 + 2] = bf2f(prow[i]);
        sJ[i * 5 + 3] = (i < 63) ? bf2f(prow[i + 1]) : 0.f;
        sJ[i * 5 + 4] = s4;
    }
    __syncthreads();
    // O = P Vh : wave owns 16(i) x 64(e)
    v8f o[4] = {};
    for (int k0 = 0; k0 < 64; k0 += 32) {
        v16bf a = frag_a_bf16(P + i0 * 72 + k0, 72);
#pragma unroll
        for (int et = 0; et < 4; ++et) {
            v16bf b = frag_b_bf16(vpT + (size_t)(h * HD + et * 16) * NTOK + tbase + k0, NTOK);
            o[et] = wmma_bf16(a, b, o[et]);
        }
    }
#pragma unroll
    for (int et = 0; et < 4; ++et) {
        int e = et * 16 + n15;
#pragma unroll
        for (int r = 0; r < 8; ++r) {
            int i = i0 + r + hi * 8;
            float w2 = sJ[i * 5 + 0] * rvS[e] + sJ[i * 5 + 1] * rvS[64 + e] +
                       sJ[i * 5 + 2] * rvS[128 + e] + sJ[i * 5 + 3] * rvS[192 + e] +
                       sJ[i * 5 + 4] * rvS[256 + e];
            xo[(size_t)(tbase + i) * DD + h * HD + e] = f2bf(o[et][r] + w2);
        }
    }
}

extern "C" void kernel_launch(void* const* d_in, const int* in_sizes, int n_in,
                              void* d_out, int out_size, void* d_ws, size_t ws_size,
                              hipStream_t stream) {
    (void)in_sizes; (void)n_in; (void)out_size; (void)ws_size;
    const float* q     = (const float*)d_in[0];
    const float* k     = (const float*)d_in[1];
    const float* v     = (const float*)d_in[2];
    const float* masks = (const float*)d_in[3];
    const float* Wq = (const float*)d_in[4];  const float* bq = (const float*)d_in[5];
    const float* Wk = (const float*)d_in[6];  const float* bk = (const float*)d_in[7];
    const float* Wv = (const float*)d_in[8];  const float* bv = (const float*)d_in[9];
    const float* Wo = (const float*)d_in[10]; const float* bo = (const float*)d_in[11];
    const float* relk = (const float*)d_in[12];
    const float* relv = (const float*)d_in[13];
    float* out = (float*)d_out;

    const size_t NE = (size_t)NTOK * DD;     // 16M elems = 32 MB bf16 each
    const size_t WE = (size_t)DD * DD;       // 256K elems = 512 KB bf16 each
    bf16* xT  = (bf16*)d_ws;
    bf16* qp  = xT + NE;
    bf16* kp  = qp + NE;
    bf16* vpT = kp + NE;
    bf16* wqb = vpT + NE;
    bf16* wkb = wqb + WE;
    bf16* wvb = wkb + WE;
    bf16* wob = wvb + WE;
    bf16* xo  = xT;  // xT consumed by projections before attention writes xo

    dim3 tgrid(TT / 32, DD / 32, BB);
    dim3 wgrid(WE / (256 * 4));              // 256 blocks
    dim3 pgridT(NTOK / 128, DD / 128);
    dim3 pgridF(DD / 128, NTOK / 128);
    dim3 agrid(NTOK / CHK, HH);

    // One-time fp32 -> bf16 weight conversion (keeps GEMM inner loops pure load+WMMA)
    k_cvt_w<<<wgrid, 256, 0, stream>>>(Wq, wqb);
    k_cvt_w<<<wgrid, 256, 0, stream>>>(Wk, wkb);
    k_cvt_w<<<wgrid, 256, 0, stream>>>(Wv, wvb);
    k_cvt_w<<<wgrid, 256, 0, stream>>>(Wo, wob);

    // Q projection (token-major)
    k_transpose_cvt<<<tgrid, 256, 0, stream>>>(q, xT);
    k_gemm_tok<<<pgridT, 256, 0, stream>>>(xT, wqb, bq, qp);
    // K projection (token-major)
    k_transpose_cvt<<<tgrid, 256, 0, stream>>>(k, xT);
    k_gemm_tok<<<pgridT, 256, 0, stream>>>(xT, wkb, bk, kp);
    // V projection (feature-major)
    k_transpose_cvt<<<tgrid, 256, 0, stream>>>(v, xT);
    k_gemm_feat<false><<<pgridF, 256, 0, stream>>>(wvb, xT, bv, nullptr, vpT, nullptr);
    // Attention (writes xo = token-major bf16, reusing xT region)
    k_attn<<<agrid, 128, 0, stream>>>(qp, kp, vpT, masks, relk, relv, xo);
    // Output projection + bias + mask -> fp32 (B, D, T)
    k_gemm_feat<true><<<pgridF, 256, 0, stream>>>(wob, xo, bo, masks, nullptr, out);
}
